// DisentangledSelfAttention_75213467287882
// MI455X (gfx1250) — compile-verified, matrix operand order checked by reading, not run on hardware
//
#include <hip/hip_runtime.h>
#include <hip/hip_bf16.h>
#include <math.h>
#include <stdint.h>

typedef __bf16 bf16;
typedef __attribute__((ext_vector_type(16))) __bf16 v16bf;
typedef __attribute__((ext_vector_type(8)))  float  v8f;

#define DEV __device__ __forceinline__

static constexpr int S_ = 1024, D_ = 1024, H_ = 16, DH_ = 64, P_ = 512;

DEV v8f zero8() { v8f z = {0.f,0.f,0.f,0.f,0.f,0.f,0.f,0.f}; return z; }

DEV bf16 u16_as_bf16(unsigned short u) {
  union { unsigned short s; bf16 b; } c; c.s = u; return c.b;
}

// Load a 16x32 bf16 fragment (A-layout; symmetric layout reused for B from
// row-major [N][K] storage). lanes 0-15: row=lane, K 0..7 & 16..23;
// lanes 16-31: row=lane-16, K 8..15 & 24..31.
DEV v16bf wmma_frag_rm(const bf16* p, int ld, int lane) {
  int row = lane & 15;
  int kb  = (lane & 16) ? 8 : 0;
  const bf16* q = p + row * ld + kb;
  v16bf f;
#pragma unroll
  for (int i = 0; i < 8; ++i) { f[i] = q[i]; f[8 + i] = q[16 + i]; }
  return f;
}

DEV v8f wmma_bf16(v16bf a, v16bf b, v8f c) {
  return __builtin_amdgcn_wmma_f32_16x16x32_bf16(false, a, false, b,
                                                 (short)0, c, false, false);
}

// ---------------------------------------------------------------------------
// Kernel 1: C[M,1024] = A[M,1024] @ W[1024,1024] + bias, output re-laid out
// head-major as bf16: out[((b*16+h)*SL + s)*64 + dh], b=row>>slShift.
// ---------------------------------------------------------------------------
__global__ __launch_bounds__(256)
void gemm_proj_kernel(const float* __restrict__ A, const float* __restrict__ W,
                      const float* __restrict__ bias, bf16* __restrict__ out,
                      int slShift) {
  __shared__ alignas(16) bf16 As[128][40];   // [m][k], BK=32 (+8 pad)
  __shared__ alignas(16) bf16 Wt[128][40];   // [n][k] (W tile stored transposed)

  const int tid = threadIdx.x, lane = tid & 31, wid = tid >> 5;
  const int wm = (wid & 3) * 32;   // 4 waves along M (32 rows each)
  const int wn = (wid >> 2) * 64;  // 2 waves along N (64 cols each)
  const int m0 = blockIdx.x * 128, n0 = blockIdx.y * 128;
  const int slMask = (1 << slShift) - 1;

  v8f acc[2][4];
#pragma unroll
  for (int i = 0; i < 2; ++i)
#pragma unroll
    for (int j = 0; j < 4; ++j) acc[i][j] = zero8();

  const int ar = tid >> 1, ac = (tid & 1) * 16;  // A: 2 threads per row
  const int wr = tid >> 3, wc = (tid & 7) * 16;  // W: 8 threads per row

  for (int kt = 0; kt < D_ / 32; ++kt) {
    __syncthreads();
    { // A tile 128x32 f32 -> bf16 LDS
      const float4* ap = (const float4*)(A + (size_t)(m0 + ar) * D_ + kt * 32 + ac);
#pragma unroll
      for (int v = 0; v < 4; ++v) {
        float4 x = ap[v];
        As[ar][ac + v * 4 + 0] = (bf16)x.x; As[ar][ac + v * 4 + 1] = (bf16)x.y;
        As[ar][ac + v * 4 + 2] = (bf16)x.z; As[ar][ac + v * 4 + 3] = (bf16)x.w;
      }
    }
    { // W tile 32x128 f32 -> transposed bf16 LDS [n][k]
      const float4* wp = (const float4*)(W + (size_t)(kt * 32 + wr) * D_ + n0 + wc);
#pragma unroll
      for (int v = 0; v < 4; ++v) {
        float4 x = wp[v];
        Wt[wc + v * 4 + 0][wr] = (bf16)x.x; Wt[wc + v * 4 + 1][wr] = (bf16)x.y;
        Wt[wc + v * 4 + 2][wr] = (bf16)x.z; Wt[wc + v * 4 + 3][wr] = (bf16)x.w;
      }
    }
    __syncthreads();

    v16bf afr[2], bfr[4];
#pragma unroll
    for (int mi = 0; mi < 2; ++mi) afr[mi] = wmma_frag_rm(&As[wm + mi * 16][0], 40, lane);
#pragma unroll
    for (int ni = 0; ni < 4; ++ni) bfr[ni] = wmma_frag_rm(&Wt[wn + ni * 16][0], 40, lane);
#pragma unroll
    for (int mi = 0; mi < 2; ++mi)
#pragma unroll
      for (int ni = 0; ni < 4; ++ni)
        acc[mi][ni] = wmma_bf16(afr[mi], bfr[ni], acc[mi][ni]);
  }

  const int nlo = lane & 15, mhalf = (lane >> 4) * 8;
#pragma unroll
  for (int mi = 0; mi < 2; ++mi)
#pragma unroll
    for (int ni = 0; ni < 4; ++ni) {
      int gn = n0 + wn + ni * 16 + nlo;
      float bv = bias[gn];
      int h = gn >> 6, dh = gn & 63;
#pragma unroll
      for (int r = 0; r < 8; ++r) {
        int gm = m0 + wm + mi * 16 + mhalf + r;
        int bb = gm >> slShift, s = gm & slMask;
        out[((((size_t)(bb * H_ + h) << slShift) + s) << 6) + dh] =
            (bf16)(acc[mi][ni][r] + bv);
      }
    }
}

// ---------------------------------------------------------------------------
// Kernel 2: rel-position tables.  out[bh, m, n] = (X[bh,m,:] . Pos[h,n,:])/scale
// X: [B*H, S, 64] bf16, Pos: [H, 512, 64] bf16, out f32 [B*H, S, 512].
// K-dim = 64 -> fragments straight from global, no LDS needed.
// ---------------------------------------------------------------------------
__global__ __launch_bounds__(256)
void gemm_rel_kernel(const bf16* __restrict__ Xb, const bf16* __restrict__ Pos,
                     float* __restrict__ out, float inv_scale) {
  const int tid = threadIdx.x, lane = tid & 31, wid = tid >> 5;
  const int m0 = blockIdx.x * 64 + (wid & 1) * 32;
  const int n0 = blockIdx.y * 128 + (wid >> 1) * 32;
  const int bh = blockIdx.z, h = bh & (H_ - 1);
  const bf16* Xbase = Xb + (size_t)bh * S_ * DH_;
  const bf16* Pbase = Pos + (size_t)h * P_ * DH_;

  v16bf a[2][2], b[2][2];
#pragma unroll
  for (int mi = 0; mi < 2; ++mi)
#pragma unroll
    for (int kc = 0; kc < 2; ++kc)
      a[mi][kc] = wmma_frag_rm(Xbase + (size_t)(m0 + mi * 16) * DH_ + kc * 32, DH_, lane);
#pragma unroll
  for (int ni = 0; ni < 2; ++ni)
#pragma unroll
    for (int kc = 0; kc < 2; ++kc)
      b[ni][kc] = wmma_frag_rm(Pbase + (size_t)(n0 + ni * 16) * DH_ + kc * 32, DH_, lane);

  const int nlo = lane & 15, mhalf = (lane >> 4) * 8;
#pragma unroll
  for (int mi = 0; mi < 2; ++mi)
#pragma unroll
    for (int ni = 0; ni < 2; ++ni) {
      v8f acc = zero8();
      acc = wmma_bf16(a[mi][0], b[ni][0], acc);
      acc = wmma_bf16(a[mi][1], b[ni][1], acc);
      int gn = n0 + ni * 16 + nlo;
#pragma unroll
      for (int r = 0; r < 8; ++r) {
        int gm = m0 + mi * 16 + mhalf + r;
        out[((size_t)bh * S_ + gm) * P_ + gn] = acc[r] * inv_scale;
      }
    }
}

// ---------------------------------------------------------------------------
// Kernel 3: flash attention with disentangled rel-position bias.
// Block = 4 waves, each wave owns 16 q-rows; loop over 64-wide K/V tiles.
// K tile streamed with gfx1250 GLOBAL_LOAD_ASYNC_TO_LDS_B128 (ASYNCcnt).
// ---------------------------------------------------------------------------
__global__ __launch_bounds__(128)
void attn_kernel(const bf16* __restrict__ Qb, const bf16* __restrict__ Kb,
                 const bf16* __restrict__ Vb, const float* __restrict__ c2p,
                 const float* __restrict__ p2c, const int* __restrict__ mask,
                 float* __restrict__ out, float inv_scale) {
  __shared__ alignas(16) int  idxtab[2048];   // packed (c2p_idx | p2c_idx<<16)
  __shared__ alignas(16) bf16 Kt[64][72];     // [k][dh]
  __shared__ alignas(16) bf16 Vt[64][72];     // transposed: [dh][k]
  __shared__ alignas(16) bf16 Pt[4][16][72];  // per-wave probs staging

  const int tid = threadIdx.x, lane = tid & 31, wid = tid >> 5;
  const int nlo = lane & 15, mhalf = (lane >> 4) * 8;
  const int bh = blockIdx.y, b = bh >> 4, h = bh & 15;
  const int q0 = blockIdx.x * 64 + wid * 16;

  // log-bucket table (replicates make_log_bucket_position), rel = i-1023
  const float kLogCoef = 127.0f / logf(511.0f / 128.0f);
  for (int i = tid; i < 2048; i += 128) {
    int r = i - 1023;
    int ar = r < 0 ? -r : r;
    int bkt, bkt2;
    if (ar <= 128) { bkt = r; bkt2 = -r; }
    else {
      float lp = ceilf(logf((float)ar * (1.0f / 128.0f)) * kLogCoef) + 128.0f;
      int l = (int)lp;
      bkt = (r > 0) ? l : -l;
      bkt2 = -bkt;
    }
    int cidx = bkt + 256;  cidx = cidx < 0 ? 0 : (cidx > 511 ? 511 : cidx);
    int pidx = -bkt2 + 256; pidx = pidx < 0 ? 0 : (pidx > 511 ? 511 : pidx);
    idxtab[i] = cidx | (pidx << 16);
  }

  const bf16* Qbase = Qb + ((size_t)bh * S_ + q0) * DH_;
  v16bf aq[2];
  aq[0] = wmma_frag_rm(Qbase, DH_, lane);
  aq[1] = wmma_frag_rm(Qbase + 32, DH_, lane);

  float m_i[8], l_i[8];
  v8f o[4];
#pragma unroll
  for (int r = 0; r < 8; ++r) { m_i[r] = -INFINITY; l_i[r] = 0.f; }
#pragma unroll
  for (int n = 0; n < 4; ++n) o[n] = zero8();

  const unsigned int* Ksrc = (const unsigned int*)(Kb + (size_t)bh * S_ * DH_);
  const unsigned int* Vsrc = (const unsigned int*)(Vb + (size_t)bh * S_ * DH_);
  const float* c2prow = c2p + (size_t)bh * S_ * P_;
  const float* p2crow = p2c + (size_t)bh * S_ * P_;

  const int ldrow = tid >> 1, ldc = (tid & 1) * 16;  // uint granularity

  for (int kt = 0; kt < S_ / 64; ++kt) {
    const int kbase = kt * 64;
    __syncthreads();
    { // K tile -> LDS via async DMA: 512 b128 chunks, 4 per thread
#pragma unroll
      for (int i = 0; i < 4; ++i) {
        int c = tid + i * 128;
        int row = c >> 3, c16 = c & 7;   // 8 x 16B chunks per 128B row
        unsigned long long ga = (unsigned long long)(uintptr_t)
            (Ksrc + (size_t)(kbase + row) * 32 + c16 * 4);
        unsigned int lo = (unsigned int)(uintptr_t)
            ((unsigned int*)&Kt[row][0] + c16 * 4);
        asm volatile("global_load_async_to_lds_b128 %0, %1, off"
                     :: "v"(lo), "v"(ga) : "memory");
      }
    }
    { // V tile -> LDS transposed [dh][k] (async copy cannot transpose)
      const uint4* src = (const uint4*)(Vsrc + (size_t)(kbase + ldrow) * 32 + ldc);
#pragma unroll
      for (int v = 0; v < 4; ++v) {
        uint4 x = src[v];
        unsigned int xs[4] = {x.x, x.y, x.z, x.w};
#pragma unroll
        for (int j = 0; j < 4; ++j) {
          int ci = ldc + v * 4 + j;
          Vt[ci * 2 + 0][ldrow] = u16_as_bf16((unsigned short)(xs[j] & 0xffff));
          Vt[ci * 2 + 1][ldrow] = u16_as_bf16((unsigned short)(xs[j] >> 16));
        }
      }
    }
    if (kt + 1 < S_ / 64) {  // -> global_prefetch_b8 for next tiles
      __builtin_prefetch((const void*)(Ksrc + (size_t)(kbase + 64 + ldrow) * 32), 0, 1);
      __builtin_prefetch((const void*)(Vsrc + (size_t)(kbase + 64 + ldrow) * 32), 0, 1);
    }
    asm volatile("s_wait_asynccnt 0x0" ::: "memory");
    __syncthreads();

    // scores = Q @ K^T  (B frag from row-major [n][dh] == A-style load)
    v8f s[4];
#pragma unroll
    for (int nt = 0; nt < 4; ++nt) {
      v16bf bk0 = wmma_frag_rm(&Kt[nt * 16][0], 72, lane);
      v16bf bk1 = wmma_frag_rm(&Kt[nt * 16][32], 72, lane);
      v8f a0 = zero8();
      a0 = wmma_bf16(aq[0], bk0, a0);
      a0 = wmma_bf16(aq[1], bk1, a0);
      s[nt] = a0;
    }

    // rel-position bias (L2-resident gathers) + mask
#pragma unroll
    for (int nt = 0; nt < 4; ++nt) {
      int gk = kbase + nt * 16 + nlo;
#pragma unroll
      for (int r = 0; r < 8; ++r) {
        int gq = q0 + mhalf + r;
        int pack = idxtab[gq - gk + 1023];
        float c2pv = c2prow[(size_t)gq * P_ + (pack & 0xffff)];
        float p2cv = p2crow[(size_t)gk * P_ + (pack >> 16)];
        int mv = mask[((size_t)b * S_ + gq) * S_ + gk];
        float val = s[nt][r] * inv_scale + c2pv + p2cv;
        s[nt][r] = mv ? val : -3.4028234663852886e38f;
      }
    }

    // online softmax (row = 16 lanes of one half; reduce via shfl_xor)
    float alpha[8];
#pragma unroll
    for (int r = 0; r < 8; ++r) {
      float t = fmaxf(fmaxf(s[0][r], s[1][r]), fmaxf(s[2][r], s[3][r]));
#pragma unroll
      for (int off = 1; off < 16; off <<= 1) t = fmaxf(t, __shfl_xor(t, off));
      float mn = fmaxf(m_i[r], t);
      alpha[r] = (m_i[r] > -1e37f) ? __expf(m_i[r] - mn) : 0.f;
      float ps = 0.f;
#pragma unroll
      for (int nt = 0; nt < 4; ++nt) {
        float p = __expf(s[nt][r] - mn);
        s[nt][r] = p; ps += p;
      }
#pragma unroll
      for (int off = 1; off < 16; off <<= 1) ps += __shfl_xor(ps, off);
      l_i[r] = l_i[r] * alpha[r] + ps;
      m_i[r] = mn;
    }
#pragma unroll
    for (int nt = 0; nt < 4; ++nt)
#pragma unroll
      for (int r = 0; r < 8; ++r) o[nt][r] *= alpha[r];

    // stage P (C layout -> A layout) through wave-private LDS
#pragma unroll
    for (int nt = 0; nt < 4; ++nt)
#pragma unroll
      for (int r = 0; r < 8; ++r)
        Pt[wid][mhalf + r][nt * 16 + nlo] = (bf16)s[nt][r];

    // O += P @ V   (B frag from transposed [dh][k] == A-style load)
#pragma unroll
    for (int kc = 0; kc < 2; ++kc) {
      v16bf ap = wmma_frag_rm(&Pt[wid][0][kc * 32], 72, lane);
#pragma unroll
      for (int nd = 0; nd < 4; ++nd) {
        v16bf bv = wmma_frag_rm(&Vt[nd * 16][kc * 32], 72, lane);
        o[nd] = wmma_bf16(ap, bv, o[nd]);
      }
    }
  }

#pragma unroll
  for (int r = 0; r < 8; ++r) {
    float inv_l = (l_i[r] > 0.f) ? 1.0f / l_i[r] : 0.f;
    int gq = q0 + mhalf + r;
#pragma unroll
    for (int nd = 0; nd < 4; ++nd)
      out[((size_t)b * S_ + gq) * D_ + h * DH_ + nd * 16 + nlo] = o[nd][r] * inv_l;
  }
}

// ---------------------------------------------------------------------------
extern "C" void kernel_launch(void* const* d_in, const int* in_sizes, int n_in,
                              void* d_out, int out_size, void* d_ws, size_t ws_size,
                              hipStream_t stream) {
  (void)in_sizes; (void)n_in; (void)out_size; (void)ws_size;
  const float* hidden = (const float*)d_in[0];
  const int*   amask  = (const int*)  d_in[1];
  const float* relemb = (const float*)d_in[2];
  const float* Wq = (const float*)d_in[3];
  const float* bq = (const float*)d_in[4];
  const float* Wk = (const float*)d_in[5];
  const float* bk = (const float*)d_in[6];
  const float* Wv = (const float*)d_in[7];
  const float* bv = (const float*)d_in[8];
  float* out = (float*)d_out;

  char* ws = (char*)d_ws;                     // workspace layout (~142 MB):
  bf16*  Qb   = (bf16*) (ws);                 //  [0,  4MB)  Q  bf16 [B,H,S,64]
  bf16*  Kbuf = (bf16*) (ws + (4ull  << 20)); //  [4,  8MB)  K
  bf16*  Vbuf = (bf16*) (ws + (8ull  << 20)); //  [8, 12MB)  V
  bf16*  PosQ = (bf16*) (ws + (12ull << 20)); //  [12,13MB)  posQ [H,512,64]
  bf16*  PosK = (bf16*) (ws + (13ull << 20)); //  [13,14MB)  posK
  float* c2p  = (float*)(ws + (14ull << 20)); //  [14,78MB)  c2p_att/scale f32
  float* p2c  = (float*)(ws + (78ull << 20)); //  [78,142MB) p2c_att/scale f32

  const float inv_scale = 0.07216878364870323f;  // 1/sqrt(64*3)

  dim3 blk(256);
  gemm_proj_kernel<<<dim3(16, 8), blk, 0, stream>>>(hidden, Wq, bq, Qb,   10);
  gemm_proj_kernel<<<dim3(16, 8), blk, 0, stream>>>(hidden, Wk, bk, Kbuf, 10);
  gemm_proj_kernel<<<dim3(16, 8), blk, 0, stream>>>(hidden, Wv, bv, Vbuf, 10);
  gemm_proj_kernel<<<dim3(4, 8),  blk, 0, stream>>>(relemb, Wq, bq, PosQ,  9);
  gemm_proj_kernel<<<dim3(4, 8),  blk, 0, stream>>>(relemb, Wk, bk, PosK,  9);

  gemm_rel_kernel<<<dim3(16, 4, 32), blk, 0, stream>>>(Qb,   PosK, c2p, inv_scale);
  gemm_rel_kernel<<<dim3(16, 4, 32), blk, 0, stream>>>(Kbuf, PosQ, p2c, inv_scale);

  attn_kernel<<<dim3(16, 32), dim3(128), 0, stream>>>(Qb, Kbuf, Vbuf, c2p, p2c,
                                                      amask, out, inv_scale);
}
